// TransformerBlock_88270167867401
// MI455X (gfx1250) — matrix operand order repaired and checked by reading, not compiled, not run on hardware
//
#include <hip/hip_runtime.h>
#include <cstdint>
#include <cstddef>

typedef __bf16 bf16_t;
typedef __attribute__((ext_vector_type(16))) __bf16 v16bf;
typedef __attribute__((ext_vector_type(8)))  __bf16 v8bf;
typedef __attribute__((ext_vector_type(8)))  float  v8f;

#define NTOK 4096
#define SCH  128
#define HEC  32
#define HIDC 160
#define DHP  64   // spatial head dim padded (real 40)
#define DCP  48   // channel head dim padded (real 40)

// ---------------- WMMA helpers (CDNA5 16x16x32 bf16, f32 accum) ----------------
static __device__ __forceinline__ v8f wmma_bf16(v16bf a, v16bf b, v8f c) {
  return __builtin_amdgcn_wmma_f32_16x16x32_bf16(false, a, false, b, (short)0, c,
                                                 false, false);
}
static __device__ __forceinline__ v8bf zero8() {
  v8bf z;
#pragma unroll
  for (int i = 0; i < 8; ++i) z[i] = (bf16_t)0.0f;
  return z;
}
static __device__ __forceinline__ v16bf zero16() {
  v16bf z;
#pragma unroll
  for (int i = 0; i < 16; ++i) z[i] = (bf16_t)0.0f;
  return z;
}
static __device__ __forceinline__ v8f zerof8() {
  v8f z;
#pragma unroll
  for (int i = 0; i < 8; ++i) z[i] = 0.0f;
  return z;
}
// A fragment (16xK tile row): lane hi group: elems 0..15 = row[hi*16+e]
static __device__ __forceinline__ v16bf load_a(const bf16_t* row, int hi) {
  v8bf lo = *(const v8bf*)(row + hi * 16);
  v8bf hh = *(const v8bf*)(row + hi * 16 + 8);
  return __builtin_shufflevector(lo, hh, 0, 1, 2, 3, 4, 5, 6, 7,
                                 8, 9, 10, 11, 12, 13, 14, 15);
}
// A fragment with upper 16 K-values zero (K-padding)
static __device__ __forceinline__ v16bf pad_a(const bf16_t* p) {
  v8bf lo = *(const v8bf*)p;
  v8bf z  = zero8();
  return __builtin_shufflevector(lo, z, 0, 1, 2, 3, 4, 5, 6, 7,
                                 8, 9, 10, 11, 12, 13, 14, 15);
}
static __device__ __forceinline__ float rmax16(float v) {
#pragma unroll
  for (int m = 1; m < 16; m <<= 1) v = fmaxf(v, __shfl_xor(v, m, 32));
  return v;
}
static __device__ __forceinline__ float rsum16(float v) {
#pragma unroll
  for (int m = 1; m < 16; m <<= 1) v += __shfl_xor(v, m, 32);
  return v;
}

// ---------------- 1) Pool / patch embed + LayerNorms ----------------
__global__ void embed_kernel(const float* __restrict__ s, const float* __restrict__ h,
                             const float* __restrict__ pos_s, const float* __restrict__ pos_h,
                             const float* __restrict__ patch_w, const float* __restrict__ patch_b,
                             const float* __restrict__ ns_g, const float* __restrict__ ns_b,
                             const float* __restrict__ nh_g, const float* __restrict__ nh_b,
                             float* __restrict__ sh, bf16_t* __restrict__ sh_bf,
                             bf16_t* __restrict__ st_bf, bf16_t* __restrict__ ht_bf) {
  int n = blockIdx.x, c = threadIdx.x;
  int d0 = n >> 8, d1 = (n >> 4) & 15, d2 = n & 15;
  int sb = ((2 * d0) * 32 + 2 * d1) * 32 + 2 * d2;
  __shared__ float red[128];
  const float* sp = s + (size_t)c * 32768 + sb;
  float val = (sp[0] + sp[1] + sp[32] + sp[33] + sp[1024] + sp[1025] + sp[1056] + sp[1057]) * 0.125f
              + pos_s[n * SCH + c];
  red[c] = val; __syncthreads();
  for (int st = 64; st > 0; st >>= 1) { if (c < st) red[c] += red[c + st]; __syncthreads(); }
  float mean = red[0] * (1.0f / 128.0f); __syncthreads();
  float dd = val - mean; red[c] = dd * dd; __syncthreads();
  for (int st = 64; st > 0; st >>= 1) { if (c < st) red[c] += red[c + st]; __syncthreads(); }
  float var = red[0] * (1.0f / 128.0f);
  float sv = dd * rsqrtf(var + 1e-5f) * ns_g[c] + ns_b[c];
  sh[(size_t)n * HIDC + c] = sv;
  sh_bf[(size_t)n * HIDC + c] = (bf16_t)sv;
  st_bf[(size_t)n * SCH + c]  = (bf16_t)sv;
  if (c < 32) {
    float acc = patch_b[c];
#pragma unroll
    for (int ci = 0; ci < 7; ++ci) {
      const float* hp = h + (size_t)ci * 32768 + sb;
      const float* w  = patch_w + (c * 7 + ci) * 8;
      acc += hp[0] * w[0] + hp[1] * w[1] + hp[32] * w[2] + hp[33] * w[3]
           + hp[1024] * w[4] + hp[1025] * w[5] + hp[1056] * w[6] + hp[1057] * w[7];
    }
    float hv = acc + pos_h[n * HEC + c];
    float s1 = hv;
#pragma unroll
    for (int m = 1; m < 32; m <<= 1) s1 += __shfl_xor(s1, m, 32);
    float m2 = s1 * (1.0f / 32.0f);
    float dv = hv - m2;
    float s2 = dv * dv;
#pragma unroll
    for (int m = 1; m < 32; m <<= 1) s2 += __shfl_xor(s2, m, 32);
    float var2 = s2 * (1.0f / 32.0f);
    float htv = dv * rsqrtf(var2 + 1e-5f) * nh_g[c] + nh_b[c];
    sh[(size_t)n * HIDC + 128 + c] = htv;
    sh_bf[(size_t)n * HIDC + 128 + c] = (bf16_t)htv;
    ht_bf[(size_t)n * HEC + c] = (bf16_t)htv;
  }
}

// ---------------- 2) Weight conversions / transposes ----------------
__global__ void cvt_kernel(const float* __restrict__ src, bf16_t* __restrict__ dst, int n) {
  int i = blockIdx.x * 256 + threadIdx.x;
  if (i < n) dst[i] = (bf16_t)src[i];
}
__global__ void convw_kernel(const float* __restrict__ cw, bf16_t* __restrict__ wT2) {
  int i = blockIdx.x * 256 + threadIdx.x;
  if (i >= 160 * 160 * 27) return;
  int co = i / (160 * 27), rem = i % (160 * 27), ci = rem / 27, off = rem % 27;
  wT2[(size_t)(off * 160 + co) * 160 + ci] = (bf16_t)cw[i];
}
__global__ void reconw_kernel(const float* __restrict__ rw, bf16_t* __restrict__ rw2) {
  int i = blockIdx.x * 256 + threadIdx.x;
  if (i >= 160 * 160 * 8) return;
  int c = i / 1280, o = (i / 8) % 160, ijk = i % 8;
  rw2[(size_t)(o * 8 + ijk) * 160 + c] = (bf16_t)rw[i];
}

// ---------------- 3) Projection GEMMs: out = X @ W^T, scatter per consumer layout ----------------
__global__ void proj_kernel(const bf16_t* __restrict__ A, const bf16_t* __restrict__ W,
                            bf16_t* __restrict__ dst, int K, int Ktiles, int mode) {
  int nt = blockIdx.x, ot = blockIdx.y;
  int lane = threadIdx.x, lr = lane & 15, hi = lane >> 4;
  const bf16_t* arow = A + (size_t)(nt * 16 + lr) * K;
  const bf16_t* wrow = W + (size_t)(ot * 16 + lr) * K;
  v8f c = zerof8();
  for (int kt = 0; kt < Ktiles; ++kt) {
    v16bf a = load_a(arow + kt * 32, hi);
    v16bf b = *(const v16bf*)(wrow + kt * 32 + hi * 16);
    c = wmma_bf16(a, b, c);
  }
  int o = ot * 16 + lr;
#pragma unroll
  for (int j = 0; j < 8; ++j) {
    int n = nt * 16 + hi * 8 + j;
    bf16_t v = (bf16_t)c[j];
    if (mode == 0) {              // q_c -> q_cT[h*32+c2][n2]  (128 x 4096)
      unsigned g = (unsigned)n * 128u + o;
      unsigned hh = g >> 17, rm = g & 131071u, n2 = rm >> 5, c2 = rm & 31u;
      dst[(size_t)(hh * 32 + c2) * 4096 + n2] = v;
    } else if (mode == 1 || mode == 2) { // q_s / k_s -> (NH,4096,64) padded
      unsigned g = (unsigned)n * 160u + o;
      unsigned hh = g / 163840u, rm = g % 163840u, n2 = rm / 40u, dd = rm % 40u;
      dst[(size_t)(hh * 4096 + n2) * DHP + dd] = v;
    } else if (mode == 3) {       // k_c -> k_cT (NH,48,4096)
      unsigned g = (unsigned)n * 160u + o;
      unsigned hh = g / 163840u, rm = g % 163840u, n2 = rm / 40u, dd = rm % 40u;
      dst[(size_t)(hh * 48 + dd) * 4096 + n2] = v;
    } else if (mode == 4) {       // v_c -> (NH,4096,48) padded
      unsigned g = (unsigned)n * 160u + o;
      unsigned hh = g / 163840u, rm = g % 163840u, n2 = rm / 40u, dd = rm % 40u;
      dst[(size_t)(hh * 4096 + n2) * DCP + dd] = v;
    } else {                      // v_s -> vT (NH,16,4096) e-major padded
      unsigned g = (unsigned)n * 32u + o;
      unsigned hh = g >> 15, rm = g & 32767u, n2 = rm >> 3, e2 = rm & 7u;
      dst[(size_t)(hh * 16 + e2) * 4096 + n2] = v;
    }
  }
}

// ---------------- 4) Channel attention scores + softmax (one wave per head) ----------------
__global__ void ca1_kernel(const bf16_t* __restrict__ qcT, const bf16_t* __restrict__ kcT,
                           const float* __restrict__ temp1, bf16_t* __restrict__ attnca) {
  int h = blockIdx.x;
  int lane = threadIdx.x, lr = lane & 15, hi = lane >> 4;
  v8f cc[2][3];
#pragma unroll
  for (int a = 0; a < 2; ++a)
#pragma unroll
    for (int b = 0; b < 3; ++b) cc[a][b] = zerof8();
  for (int kt = 0; kt < 128; ++kt) {
    v16bf av[2], bv[3];
#pragma unroll
    for (int mt = 0; mt < 2; ++mt)
      av[mt] = load_a(qcT + (size_t)(h * 32 + mt * 16 + lr) * 4096 + kt * 32, hi);
#pragma unroll
    for (int dt = 0; dt < 3; ++dt)
      bv[dt] = *(const v16bf*)(kcT + (size_t)(h * 48 + dt * 16 + lr) * 4096 + kt * 32 + hi * 16);
#pragma unroll
    for (int mt = 0; mt < 2; ++mt)
#pragma unroll
      for (int dt = 0; dt < 3; ++dt) cc[mt][dt] = wmma_bf16(av[mt], bv[dt], cc[mt][dt]);
  }
  float sc = rsqrtf(40.0f) * temp1[h];
#pragma unroll
  for (int mt = 0; mt < 2; ++mt) {
#pragma unroll
    for (int j = 0; j < 8; ++j) {
      float s0 = cc[mt][0][j] * sc;
      float s1 = cc[mt][1][j] * sc;
      float s2 = (lr < 8) ? cc[mt][2][j] * sc : -__builtin_inff();
      float mx = rmax16(fmaxf(fmaxf(s0, s1), s2));
      float p0 = __expf(s0 - mx), p1 = __expf(s1 - mx);
      float p2 = (lr < 8) ? __expf(s2 - mx) : 0.0f;
      float inv = 1.0f / rsum16(p0 + p1 + p2);
      size_t rb = (size_t)(h * 32 + mt * 16 + hi * 8 + j) * DCP;
      attnca[rb + lr]      = (bf16_t)(p0 * inv);
      attnca[rb + 16 + lr] = (bf16_t)(p1 * inv);
      attnca[rb + 32 + lr] = (bf16_t)(p2 * inv);
    }
  }
}

// ---------------- 5) Channel attention @ V ----------------
__global__ void ca2_kernel(const bf16_t* __restrict__ attnca, const bf16_t* __restrict__ vch,
                           float* __restrict__ x) {
  int tt = blockIdx.x, h = blockIdx.y;
  int lane = threadIdx.x, lr = lane & 15, hi = lane >> 4;
  const bf16_t* vrow = vch + (size_t)((h * NTOK) + tt * 16 + lr) * DCP;
  v16bf b0 = *(const v16bf*)(vrow + hi * 16);
  v16bf b1 = (hi == 0) ? *(const v16bf*)(vrow + 32) : zero16();
#pragma unroll
  for (int mt = 0; mt < 2; ++mt) {
    const bf16_t* ar = attnca + (size_t)((h * 32) + mt * 16 + lr) * DCP;
    v16bf a0 = load_a(ar, hi);
    v16bf a1 = pad_a(ar + 32 + hi * 8);
    v8f c = zerof8();
    c = wmma_bf16(a0, b0, c);
    c = wmma_bf16(a1, b1, c);
#pragma unroll
    for (int j = 0; j < 8; ++j) {
      int cr = mt * 16 + hi * 8 + j, tok = tt * 16 + lr;
      unsigned f = (unsigned)(h * 32 + cr) * 4096u + tok;
      unsigned np = f >> 7, cp = f & 127u;
      x[(size_t)np * HIDC + cp] = c[j];
    }
  }
}

// ---------------- 6) Flash spatial attention (one wave per 16-query tile per head) ----------------
__global__ void sa_kernel(const bf16_t* __restrict__ qsH, const bf16_t* __restrict__ ksH,
                          const bf16_t* __restrict__ vT, const float* __restrict__ temp2,
                          float* __restrict__ x) {
  int qt = blockIdx.x, h = blockIdx.y;
  int lane = threadIdx.x, lr = lane & 15, hi = lane >> 4;
  float sc = rsqrtf(40.0f) * temp2[h];
  const bf16_t* qrow = qsH + (size_t)((h * NTOK) + qt * 16 + lr) * DHP;
  v16bf a0 = load_a(qrow, hi);
  v16bf a1 = load_a(qrow + 32, hi);
  float mrow[8], lrow[8];
#pragma unroll
  for (int j = 0; j < 8; ++j) { mrow[j] = -__builtin_inff(); lrow[j] = 0.0f; }
  v8f o = zerof8();
  __shared__ bf16_t pl[256];
  for (int mt = 0; mt < 256; ++mt) {
    const bf16_t* kb = ksH + (size_t)((h * NTOK) + mt * 16 + lr) * DHP;
    if (mt + 1 < 256) __builtin_prefetch(kb + 16 * DHP, 0, 1);
    v16bf b0 = *(const v16bf*)(kb + hi * 16);
    v16bf b1 = *(const v16bf*)(kb + 32 + hi * 16);
    v8f s = zerof8();
    s = wmma_bf16(a0, b0, s);
    s = wmma_bf16(a1, b1, s);
    float p[8];
#pragma unroll
    for (int j = 0; j < 8; ++j) {
      float v  = s[j] * sc;
      float mx = rmax16(v);
      float nm = fmaxf(mrow[j], mx);
      float al = __expf(mrow[j] - nm);
      float pv = __expf(v - nm);
      float rs = rsum16(pv);
      lrow[j] = lrow[j] * al + rs;
      mrow[j] = nm;
      o[j] *= al;
      p[j] = pv;
    }
#pragma unroll
    for (int j = 0; j < 8; ++j) pl[(hi * 8 + j) * 16 + lr] = (bf16_t)p[j];
    asm volatile("s_wait_dscnt 0x0" ::: "memory");
    v16bf pa = (hi == 0) ? __builtin_shufflevector(*(const v8bf*)&pl[lr * 16], *(const v8bf*)&pl[lr * 16 + 8], 0, 1, 2, 3, 4, 5, 6, 7, 8, 9, 10, 11, 12, 13, 14, 15) : zero16();
    v16bf vb = (hi == 0) ? *(const v16bf*)(vT + (size_t)((h * 16) + lr) * NTOK + mt * 16)
                         : zero16();
    o = wmma_bf16(pa, vb, o);
    asm volatile("" ::: "memory");
  }
  if (lr < 8) {
#pragma unroll
    for (int j = 0; j < 8; ++j) {
      int tok = qt * 16 + hi * 8 + j;
      unsigned f = (unsigned)(h * NTOK + tok) * 8u + lr;
      unsigned np = f >> 5, cp = f & 31u;
      x[(size_t)np * HIDC + 128 + cp] = o[j] / lrow[j];
    }
  }
}

// ---------------- 7) Residual + gamma, to bf16 voxel-major ----------------
__global__ void assemble_kernel(const float* __restrict__ sh, const float* __restrict__ x,
                                const float* __restrict__ gamma, bf16_t* __restrict__ yb) {
  int n = blockIdx.x, c = threadIdx.x;
  float a = sh[(size_t)n * HIDC + c] + x[(size_t)n * HIDC + c] * gamma[c];
  yb[(size_t)n * HIDC + c] = (bf16_t)a;
}

// ---------------- 8) 3x3x3 conv (implicit GEMM) + BN + LeakyReLU ----------------
__global__ void conv_kernel(const bf16_t* __restrict__ yb, const bf16_t* __restrict__ wT2,
                            const float* __restrict__ conv_b, const float* __restrict__ bn_g,
                            const float* __restrict__ bn_b, const float* __restrict__ bn_m,
                            const float* __restrict__ bn_v, bf16_t* __restrict__ zb) {
  int vt = blockIdx.x, ct = blockIdx.y;
  int lane = threadIdx.x, lr = lane & 15, hi = lane >> 4;
  int vox = vt * 16 + lr;
  int d0 = vox >> 8, d1 = (vox >> 4) & 15, d2 = vox & 15;
  v8f acc = zerof8();
#pragma unroll 1
  for (int off = 0; off < 27; ++off) {
    int dx = off / 9 - 1, dy = (off / 3) % 3 - 1, dz = off % 3 - 1;
    int nx = d0 + dx, ny = d1 + dy, nz = d2 + dz;
    bool ok = ((unsigned)nx < 16u) && ((unsigned)ny < 16u) && ((unsigned)nz < 16u);
    const bf16_t* row = yb + (size_t)(((nx * 16) + ny) * 16 + nz) * HIDC;
#pragma unroll
    for (int kt = 0; kt < 5; ++kt) {
      v16bf a = ok ? load_a(row + kt * 32, hi) : zero16();
      v16bf b = *(const v16bf*)(wT2 + (size_t)(off * 160 + ct * 16 + lr) * 160 + kt * 32 + hi * 16);
      acc = wmma_bf16(a, b, acc);
    }
  }
  int co = ct * 16 + lr;
  float inv = bn_g[co] * rsqrtf(bn_v[co] + 1e-5f);
  float bb = bn_b[co], bm = bn_m[co], cb = conv_b[co];
#pragma unroll
  for (int j = 0; j < 8; ++j) {
    int v2 = vt * 16 + hi * 8 + j;
    float val = acc[j] + cb;
    val = (val - bm) * inv + bb;
    val = val >= 0.0f ? val : 0.01f * val;
    zb[(size_t)v2 * HIDC + co] = (bf16_t)val;
  }
}

// ---------------- 9) Transposed conv (un-patch) GEMM ----------------
__global__ void recon_kernel(const bf16_t* __restrict__ zb, const bf16_t* __restrict__ rw2,
                             const float* __restrict__ recon_b, float* __restrict__ out) {
  int vt = blockIdx.x, nt = blockIdx.y;
  int lane = threadIdx.x, lr = lane & 15, hi = lane >> 4;
  const bf16_t* arow = zb + (size_t)(vt * 16 + lr) * HIDC;
  const bf16_t* brow = rw2 + (size_t)(nt * 16 + lr) * HIDC;
  v8f c = zerof8();
#pragma unroll
  for (int kt = 0; kt < 5; ++kt) {
    v16bf a = load_a(arow + kt * 32, hi);
    v16bf b = *(const v16bf*)(brow + kt * 32 + hi * 16);
    c = wmma_bf16(a, b, c);
  }
  int ng = nt * 16 + lr;
  int o = ng >> 3, ijk = ng & 7;
  int ii = (ijk >> 2) & 1, jj = (ijk >> 1) & 1, kk = ijk & 1;
  float rb = recon_b[o];
#pragma unroll
  for (int j = 0; j < 8; ++j) {
    int vox = vt * 16 + hi * 8 + j;
    int d0 = vox >> 8, d1 = (vox >> 4) & 15, d2 = vox & 15;
    out[(size_t)o * 32768 + (2 * d0 + ii) * 1024 + (2 * d1 + jj) * 32 + (2 * d2 + kk)] = c[j] + rb;
  }
}

// ---------------- launcher ----------------
extern "C" void kernel_launch(void* const* d_in, const int* in_sizes, int n_in,
                              void* d_out, int out_size, void* d_ws, size_t ws_size,
                              hipStream_t stream) {
  (void)in_sizes; (void)n_in; (void)out_size; (void)ws_size;
  const float* s       = (const float*)d_in[0];
  const float* hin     = (const float*)d_in[1];
  const float* pos_s   = (const float*)d_in[2];
  const float* pos_h   = (const float*)d_in[3];
  const float* patch_w = (const float*)d_in[4];
  const float* patch_b = (const float*)d_in[5];
  const float* ns_g    = (const float*)d_in[6];
  const float* ns_b    = (const float*)d_in[7];
  const float* nh_g    = (const float*)d_in[8];
  const float* nh_b    = (const float*)d_in[9];
  const float* gamma   = (const float*)d_in[10];
  const float* temp1   = (const float*)d_in[11];
  const float* temp2   = (const float*)d_in[12];
  const float* qc_w    = (const float*)d_in[13];
  const float* qs_w    = (const float*)d_in[14];
  const float* kc_w    = (const float*)d_in[15];
  const float* vc_w    = (const float*)d_in[16];
  const float* ks_w    = (const float*)d_in[17];
  const float* vs_w    = (const float*)d_in[18];
  const float* conv_w  = (const float*)d_in[19];
  const float* conv_b  = (const float*)d_in[20];
  const float* bn_g    = (const float*)d_in[21];
  const float* bn_b    = (const float*)d_in[22];
  const float* bn_m    = (const float*)d_in[23];
  const float* bn_v    = (const float*)d_in[24];
  const float* recon_w = (const float*)d_in[25];
  const float* recon_b = (const float*)d_in[26];

  char* base = (char*)d_ws;
  size_t off = 0;
  auto alloc = [&](size_t bytes) -> void* {
    void* p = base + off;
    off += (bytes + 255) & ~(size_t)255;
    return p;
  };
  float*  sh    = (float*)alloc((size_t)NTOK * HIDC * 4);
  float*  x     = (float*)alloc((size_t)NTOK * HIDC * 4);
  bf16_t* sh_bf = (bf16_t*)alloc((size_t)NTOK * HIDC * 2);
  bf16_t* st_bf = (bf16_t*)alloc((size_t)NTOK * SCH * 2);
  bf16_t* ht_bf = (bf16_t*)alloc((size_t)NTOK * HEC * 2);
  bf16_t* wqc   = (bf16_t*)alloc(128 * 128 * 2);
  bf16_t* wqs   = (bf16_t*)alloc(160 * 160 * 2);
  bf16_t* wkc   = (bf16_t*)alloc(160 * 160 * 2);
  bf16_t* wvc   = (bf16_t*)alloc(160 * 160 * 2);
  bf16_t* wks   = (bf16_t*)alloc(160 * 160 * 2);
  bf16_t* wvs   = (bf16_t*)alloc(32 * 32 * 2);
  bf16_t* qcT   = (bf16_t*)alloc((size_t)128 * 4096 * 2);
  bf16_t* attnca= (bf16_t*)alloc((size_t)4 * 32 * DCP * 2);
  bf16_t* yb    = (bf16_t*)alloc((size_t)NTOK * HIDC * 2);
  bf16_t* zb    = (bf16_t*)alloc((size_t)NTOK * HIDC * 2);
  bf16_t* wT2   = (bf16_t*)alloc((size_t)27 * 160 * 160 * 2);
  bf16_t* rw2   = (bf16_t*)alloc((size_t)1280 * 160 * 2);
  size_t zoff = off;
  bf16_t* qsH = (bf16_t*)alloc((size_t)4 * 4096 * DHP * 2);
  bf16_t* ksH = (bf16_t*)alloc((size_t)4 * 4096 * DHP * 2);
  bf16_t* kcT = (bf16_t*)alloc((size_t)4 * 48 * 4096 * 2);
  bf16_t* vch = (bf16_t*)alloc((size_t)4 * 4096 * DCP * 2);
  bf16_t* vT  = (bf16_t*)alloc((size_t)4 * 16 * 4096 * 2);
  size_t zlen = off - zoff;
  hipMemsetAsync(base + zoff, 0, zlen, stream);

  embed_kernel<<<NTOK, 128, 0, stream>>>(s, hin, pos_s, pos_h, patch_w, patch_b,
                                         ns_g, ns_b, nh_g, nh_b, sh, sh_bf, st_bf, ht_bf);
  cvt_kernel<<<(16384 + 255) / 256, 256, 0, stream>>>(qc_w, wqc, 16384);
  cvt_kernel<<<(25600 + 255) / 256, 256, 0, stream>>>(qs_w, wqs, 25600);
  cvt_kernel<<<(25600 + 255) / 256, 256, 0, stream>>>(kc_w, wkc, 25600);
  cvt_kernel<<<(25600 + 255) / 256, 256, 0, stream>>>(vc_w, wvc, 25600);
  cvt_kernel<<<(25600 + 255) / 256, 256, 0, stream>>>(ks_w, wks, 25600);
  cvt_kernel<<<(1024 + 255) / 256, 256, 0, stream>>>(vs_w, wvs, 1024);
  convw_kernel<<<(691200 + 255) / 256, 256, 0, stream>>>(conv_w, wT2);
  reconw_kernel<<<(204800 + 255) / 256, 256, 0, stream>>>(recon_w, rw2);

  proj_kernel<<<dim3(256, 8),  32, 0, stream>>>(st_bf, wqc, qcT, 128, 4, 0);
  proj_kernel<<<dim3(256, 10), 32, 0, stream>>>(sh_bf, wqs, qsH, 160, 5, 1);
  proj_kernel<<<dim3(256, 10), 32, 0, stream>>>(sh_bf, wks, ksH, 160, 5, 2);
  proj_kernel<<<dim3(256, 10), 32, 0, stream>>>(sh_bf, wkc, kcT, 160, 5, 3);
  proj_kernel<<<dim3(256, 10), 32, 0, stream>>>(sh_bf, wvc, vch, 160, 5, 4);
  proj_kernel<<<dim3(256, 2),  32, 0, stream>>>(ht_bf, wvs, vT, 32, 1, 5);

  ca1_kernel<<<4, 32, 0, stream>>>(qcT, kcT, temp1, attnca);
  ca2_kernel<<<dim3(256, 4), 32, 0, stream>>>(attnca, vch, x);
  sa_kernel<<<dim3(256, 4), 32, 0, stream>>>(qsH, ksH, vT, temp2, x);

  assemble_kernel<<<NTOK, HIDC, 0, stream>>>(sh, x, gamma, yb);
  conv_kernel<<<dim3(256, 10), 32, 0, stream>>>(yb, wT2, conv_b, bn_g, bn_b, bn_m, bn_v, zb);
  recon_kernel<<<dim3(256, 80), 32, 0, stream>>>(zb, rw2, recon_b, (float*)d_out);
}
